// SelfAttention2_86199993631019
// MI455X (gfx1250) — compile-verified, matrix operand order checked by reading, not run on hardware
//
#include <hip/hip_runtime.h>

// ---------------------------------------------------------------------------
// MI455X (gfx1250, wave32) self-attention:
//   qkv = x @ W^T  -> split on sequence dim -> q,k,v
//   S = q k^T / 32 ; P = softmax(S) ; out = P v
// GEMMs: bf16 x bf16 -> f32 via v_wmma_f32_16x16x32_bf16, operands staged
// through LDS with global_load_async_to_lds_b128 (ASYNCcnt double-buffer).
// ---------------------------------------------------------------------------

typedef __bf16 bf16_t;
typedef bf16_t v16bf __attribute__((ext_vector_type(16)));
typedef bf16_t v8bf  __attribute__((ext_vector_type(8)));
typedef float  v8f   __attribute__((ext_vector_type(8)));

__device__ __forceinline__ unsigned int f2bs(float f) {
  bf16_t h = (bf16_t)f;
  return (unsigned int)__builtin_bit_cast(unsigned short, h);
}

// Async 16B copy global -> LDS (no VGPR round trip, tracked by ASYNCcnt).
__device__ __forceinline__ void async_copy_b128(unsigned lds_off, const void* gaddr) {
  asm volatile("global_load_async_to_lds_b128 %0, %1, off"
               :: "v"(lds_off), "v"(gaddr)
               : "memory");
}

// Build one 16x32 bf16 WMMA operand fragment from an LDS panel.
// Panel layout: 128 rows x 32 bf16 (64B per row), row-major.
// lanes 0-15  : row = row0+lane,    K chunks [0..7] and [16..23]
// lanes 16-31 : row = row0+lane-16, K chunks [8..15] and [24..31]
__device__ __forceinline__ v16bf lds_frag(const char* base, int row0, int lane) {
  const int lr = lane & 15;
  const int hh = lane >> 4;
  const v8bf* p = (const v8bf*)(base + (row0 + lr) * 64 + hh * 16);
  v8bf lo = p[0];
  v8bf hi = p[2];  // +32 bytes
  return __builtin_shufflevector(lo, hi, 0, 1, 2, 3, 4, 5, 6, 7,
                                 8, 9, 10, 11, 12, 13, 14, 15);
}

// 128x128 output tile per 256-thread block; 8 waves in a 2x4 grid,
// each wave owns 64(M) x 32(N) = 4x2 WMMA accumulators.
// A/B panels (128x32 bf16 each) double-buffered in LDS via async DMA.
// MODE 0: store bf16 row-major [m][n]
// MODE 1: store f32  row-major [m][n], scaled
// MODE 2: store bf16 transposed [n][m] (packed b128 stores)
template <int MODE>
__device__ __forceinline__ void gemm128(const bf16_t* __restrict__ A, int lda,
                                        const bf16_t* __restrict__ Bt, int ldb,
                                        void* __restrict__ Cv, int ldc,
                                        int K, float scale) {
  __shared__ __align__(16) char smem[2][2][8192];  // [buf][A|B][128*32*2B]

  const int tid  = threadIdx.x;
  const int lane = tid & 31;
  const int wid  = tid >> 5;
  const int wm   = (wid >> 2) * 64;  // wave M offset inside 128-tile
  const int wn   = (wid & 3) * 32;   // wave N offset inside 128-tile

  const char* gA = (const char*)A;
  const char* gB = (const char*)Bt;

  // Thread tid copies 16B chunks {tid, tid+256} of each 8KB panel.
  // Chunk c <-> (row = c>>2, kchunk = c&3).
  auto issue_panels = [&](int buf, int k0) {
#pragma unroll
    for (int h = 0; h < 2; h++) {
      const int c   = tid + h * 256;
      const int row = c >> 2;
      const int kc  = c & 3;
      const unsigned la = (unsigned)(size_t)&smem[buf][0][c * 16];
      const unsigned lb = (unsigned)(size_t)&smem[buf][1][c * 16];
      async_copy_b128(la, gA + ((size_t)row * (size_t)lda + (size_t)(k0 + kc * 8)) * 2);
      async_copy_b128(lb, gB + ((size_t)row * (size_t)ldb + (size_t)(k0 + kc * 8)) * 2);
    }
  };

  v8f acc[4][2] = {};

  issue_panels(0, 0);
  int buf = 0;
  for (int k0 = 0; k0 < K; k0 += 32) {
    const bool has_next = (k0 + 32) < K;
    if (has_next) issue_panels(buf ^ 1, k0 + 32);
    // L2 pre-warm two K-steps ahead (global_prefetch_b8).
    if (k0 + 64 < K) {
      const int row = tid >> 1;  // 128 rows covered by 256 threads
      __builtin_prefetch(gA + ((size_t)row * (size_t)lda + (size_t)(k0 + 64)) * 2, 0, 1);
      __builtin_prefetch(gB + ((size_t)row * (size_t)ldb + (size_t)(k0 + 64)) * 2, 0, 1);
    }
    // Async transfers complete in order: after issuing 4 new ones, waiting
    // for <=4 outstanding guarantees the current panel has fully landed.
    if (has_next) asm volatile("s_wait_asynccnt 0x4" ::: "memory");
    else          asm volatile("s_wait_asynccnt 0x0" ::: "memory");
    __syncthreads();  // make all waves' DMA results visible

    v16bf a[4];
    v16bf b[2];
#pragma unroll
    for (int i = 0; i < 4; i++) a[i] = lds_frag(&smem[buf][0][0], wm + 16 * i, lane);
#pragma unroll
    for (int j = 0; j < 2; j++) b[j] = lds_frag(&smem[buf][1][0], wn + 16 * j, lane);
#pragma unroll
    for (int i = 0; i < 4; i++)
#pragma unroll
      for (int j = 0; j < 2; j++)
        acc[i][j] = __builtin_amdgcn_wmma_f32_16x16x32_bf16(
            /*neg_a=*/false, a[i], /*neg_b=*/false, b[j],
            /*c_mod=*/(short)0, acc[i][j],
            /*reuse_a=*/false, /*reuse_b=*/false);

    __syncthreads();  // protect this buffer from the next async writes
    buf ^= 1;
  }

  const int lr = lane & 15;
  const int hh = lane >> 4;

  if constexpr (MODE == 0) {
    bf16_t* C = (bf16_t*)Cv;
#pragma unroll
    for (int i = 0; i < 4; i++)
#pragma unroll
      for (int j = 0; j < 2; j++) {
        const int mb = wm + 16 * i + 8 * hh;
        const int n  = wn + 16 * j + lr;
#pragma unroll
        for (int e = 0; e < 8; e++)
          C[(size_t)(mb + e) * (size_t)ldc + n] = (bf16_t)acc[i][j][e];
      }
  } else if constexpr (MODE == 1) {
    float* C = (float*)Cv;
#pragma unroll
    for (int i = 0; i < 4; i++)
#pragma unroll
      for (int j = 0; j < 2; j++) {
        const int mb = wm + 16 * i + 8 * hh;
        const int n  = wn + 16 * j + lr;
#pragma unroll
        for (int e = 0; e < 8; e++)
          C[(size_t)(mb + e) * (size_t)ldc + n] = acc[i][j][e] * scale;
      }
  } else {  // MODE 2: transposed bf16 store; consecutive M is contiguous
    bf16_t* C = (bf16_t*)Cv;
#pragma unroll
    for (int i = 0; i < 4; i++)
#pragma unroll
      for (int j = 0; j < 2; j++) {
        const int mb = wm + 16 * i + 8 * hh;
        const int n  = wn + 16 * j + lr;
        uint4 pk;
        pk.x = f2bs(acc[i][j][0]) | (f2bs(acc[i][j][1]) << 16);
        pk.y = f2bs(acc[i][j][2]) | (f2bs(acc[i][j][3]) << 16);
        pk.z = f2bs(acc[i][j][4]) | (f2bs(acc[i][j][5]) << 16);
        pk.w = f2bs(acc[i][j][6]) | (f2bs(acc[i][j][7]) << 16);
        *(uint4*)(C + (size_t)n * (size_t)ldc + mb) = pk;
      }
  }
}

// ---------------------------------------------------------------------------

__global__ void __launch_bounds__(256)
cvt_f32_to_bf16_kernel(const float* __restrict__ in, bf16_t* __restrict__ out, int n4) {
  const int i = blockIdx.x * blockDim.x + threadIdx.x;
  if (i >= n4) return;
  const float4 v = ((const float4*)in)[i];
  uint2 p;
  p.x = f2bs(v.x) | (f2bs(v.y) << 16);
  p.y = f2bs(v.z) | (f2bs(v.w) << 16);
  ((uint2*)out)[i] = p;
}

// qkv = X[b] @ W^T. Row tiles 0..15 -> Q,K rows (row-major bf16);
// row tiles 16..23 -> V stored transposed (V^T, [feature][token]).
__global__ void __launch_bounds__(256)
qkv_gemm_kernel(const bf16_t* __restrict__ Xb, const bf16_t* __restrict__ Wb,
                bf16_t* __restrict__ QK, bf16_t* __restrict__ Vt) {
  const int mt = blockIdx.x;  // 0..23
  const int nt = blockIdx.y;  // 0..23
  const int b  = blockIdx.z;  // 0..3
  const bf16_t* A  = Xb + (size_t)b * 3072 * 1024 + (size_t)mt * 128 * 1024;
  const bf16_t* Bt = Wb + (size_t)nt * 128 * 1024;  // W rows == rows of B^T
  if (mt < 16) {
    bf16_t* C = QK + (size_t)b * 2048 * 3072 + (size_t)mt * 128 * 3072 + (size_t)nt * 128;
    gemm128<0>(A, 1024, Bt, 1024, C, 3072, 1024, 1.0f);
  } else {
    bf16_t* C = Vt + (size_t)b * 3072 * 1024 + (size_t)nt * 128 * 1024 + (size_t)(mt - 16) * 128;
    gemm128<2>(A, 1024, Bt, 1024, C, 1024, 1024, 1.0f);
  }
}

// S = Q K^T * (1/sqrt(1024)); B-operand rows are K's rows (row-major).
__global__ void __launch_bounds__(256)
scores_gemm_kernel(const bf16_t* __restrict__ QK, float* __restrict__ S) {
  const int mt = blockIdx.x;  // 0..7
  const int nt = blockIdx.y;  // 0..7
  const int b  = blockIdx.z;
  const bf16_t* Qb = QK + (size_t)b * 2048 * 3072;
  const bf16_t* A  = Qb + (size_t)mt * 128 * 3072;
  const bf16_t* Bt = Qb + (size_t)(1024 + nt * 128) * 3072;
  float* C = S + (size_t)b * 1024 * 1024 + (size_t)mt * 128 * 1024 + (size_t)nt * 128;
  gemm128<1>(A, 3072, Bt, 3072, C, 1024, 3072, 0.03125f);
}

// Row softmax: one block per row; wave32 shuffle + LDS cross-wave reduce.
__global__ void __launch_bounds__(256)
softmax_kernel(const float* __restrict__ S, bf16_t* __restrict__ P) {
  const int row = blockIdx.x;           // 0..4095 (b*1024 + m)
  const int tid = threadIdx.x;          // 0..255, 4 cols each
  const int lane = tid & 31;
  const int wid  = tid >> 5;
  const float4 v = ((const float4*)(S + (size_t)row * 1024))[tid];

  __shared__ float red_max[8];
  __shared__ float red_sum[8];

  float m = fmaxf(fmaxf(v.x, v.y), fmaxf(v.z, v.w));
#pragma unroll
  for (int o = 16; o > 0; o >>= 1) m = fmaxf(m, __shfl_xor(m, o, 32));
  if (lane == 0) red_max[wid] = m;
  __syncthreads();
  float M = red_max[0];
#pragma unroll
  for (int i = 1; i < 8; i++) M = fmaxf(M, red_max[i]);

  const float e0 = __expf(v.x - M);
  const float e1 = __expf(v.y - M);
  const float e2 = __expf(v.z - M);
  const float e3 = __expf(v.w - M);
  float s = (e0 + e1) + (e2 + e3);
#pragma unroll
  for (int o = 16; o > 0; o >>= 1) s += __shfl_xor(s, o, 32);
  if (lane == 0) red_sum[wid] = s;
  __syncthreads();
  float T = 0.0f;
#pragma unroll
  for (int i = 0; i < 8; i++) T += red_sum[i];
  const float r = 1.0f / T;

  uint2 p;
  p.x = f2bs(e0 * r) | (f2bs(e1 * r) << 16);
  p.y = f2bs(e2 * r) | (f2bs(e3 * r) << 16);
  ((uint2*)(P + (size_t)row * 1024))[tid] = p;
}

// out = P @ V, using V^T rows as B-operand (contiguous token axis).
__global__ void __launch_bounds__(256)
out_gemm_kernel(const bf16_t* __restrict__ P, const bf16_t* __restrict__ Vt,
                float* __restrict__ O) {
  const int mt = blockIdx.x;  // 0..7
  const int nt = blockIdx.y;  // 0..23
  const int b  = blockIdx.z;
  const bf16_t* A  = P + (size_t)b * 1024 * 1024 + (size_t)mt * 128 * 1024;
  const bf16_t* Bt = Vt + (size_t)b * 3072 * 1024 + (size_t)nt * 128 * 1024;
  float* C = O + (size_t)b * 1024 * 3072 + (size_t)mt * 128 * 3072 + (size_t)nt * 128;
  gemm128<1>(A, 1024, Bt, 1024, C, 3072, 1024, 1.0f);
}

// ---------------------------------------------------------------------------

extern "C" void kernel_launch(void* const* d_in, const int* in_sizes, int n_in,
                              void* d_out, int out_size, void* d_ws, size_t ws_size,
                              hipStream_t stream) {
  (void)in_sizes; (void)n_in; (void)out_size; (void)ws_size;
  const float* x = (const float*)d_in[0];   // [4,3072,1024] f32
  const float* W = (const float*)d_in[1];   // [3072,1024]  f32
  float* out = (float*)d_out;               // [4,1024,3072] f32

  // Workspace layout (bytes), total ~132.1 MB:
  char* ws = (char*)d_ws;
  bf16_t* Xb = (bf16_t*)(ws + 0);           // 4*3072*1024 bf16 = 25165824 B
  bf16_t* Wb = (bf16_t*)(ws + 25165824);    // 3072*1024  bf16 =  6291456 B
  bf16_t* QK = (bf16_t*)(ws + 31457280);    // 4*2048*3072 bf16 = 50331648 B
  bf16_t* Vt = (bf16_t*)(ws + 81788928);    // 4*3072*1024 bf16 = 25165824 B
  float*  S  = (float*) (ws + 106954752);   // 4*1024*1024 f32  = 16777216 B
  bf16_t* P  = (bf16_t*)(ws + 123731968);   // 4*1024*1024 bf16 =  8388608 B

  // 1) f32 -> bf16 conversion of inputs
  cvt_f32_to_bf16_kernel<<<dim3(3145728 / 256), 256, 0, stream>>>(x, Xb, 3145728);
  cvt_f32_to_bf16_kernel<<<dim3(786432 / 256), 256, 0, stream>>>(W, Wb, 786432);

  // 2) QKV projection (Q,K row-major; V transposed)
  qkv_gemm_kernel<<<dim3(24, 24, 4), 256, 0, stream>>>(Xb, Wb, QK, Vt);

  // 3) scores = Q K^T / 32
  scores_gemm_kernel<<<dim3(8, 8, 4), 256, 0, stream>>>(QK, S);

  // 4) softmax -> bf16 probabilities
  softmax_kernel<<<dim3(4096), 256, 0, stream>>>(S, P);

  // 5) out = P V
  out_gemm_kernel<<<dim3(8, 24, 4), 256, 0, stream>>>(P, Vt, out);
}
